// MultiHeadAttention_51659866636389
// MI455X (gfx1250) — matrix operand order repaired and checked by reading, not compiled, not run on hardware
//
#include <hip/hip_runtime.h>
#include <hip/hip_bf16.h>
#include <math.h>

// ---------------------------------------------------------------------------
// MI455X (gfx1250) fused MHA: bf16 WMMA (f32 accum) + TDM tensor_load_to_lds.
//   S=2048, H=2048, heads=16, D=128, causal, RoPE(theta=1e4)
// ---------------------------------------------------------------------------

typedef __bf16 bf16;
typedef __attribute__((ext_vector_type(16))) __bf16    v16bf;
typedef __attribute__((ext_vector_type(8)))  __bf16    v8bf;
typedef __attribute__((ext_vector_type(8)))  float     v8f;
typedef __attribute__((ext_vector_type(4)))  float     v4f;
typedef __attribute__((ext_vector_type(4)))  unsigned  v4u;
typedef __attribute__((ext_vector_type(8)))  unsigned  v8u;

#define S_LEN 2048
#define HID   2048
#define NH    16
#define DH    128
#define O3    6144   // 3*HID

__device__ __forceinline__ v16bf join16(v8bf lo, v8bf hi) {
    union { v16bf v; v8bf h[2]; } u;
    u.h[0] = lo; u.h[1] = hi;
    return u.v;
}

// A-operand fragment (16x32 bf16): lane<16 -> K {0..7},{16..23}; lane>=16 -> +8
__device__ __forceinline__ v16bf load_afrag(const bf16* row_base, int half) {
    v8bf lo = *(const v8bf*)(row_base + half * 8);
    v8bf hi = *(const v8bf*)(row_base + 16 + half * 8);
    return join16(lo, hi);
}
// B-operand fragment (32x16 bf16): lane<16 -> K {0..7},{16..23} of col n; lane>=16 -> +8
__device__ __forceinline__ v16bf load_bfrag(const bf16* col_base, int half) {
    v8bf lo = *(const v8bf*)(col_base + half * 8);
    v8bf hi = *(const v8bf*)(col_base + 16 + half * 8);
    return join16(lo, hi);
}

__device__ __forceinline__ v8f wmma_bf16(v16bf a, v16bf b, v8f c) {
    // (neg_a, A, neg_b, B, c_mod, C, reuse_a, reuse_b)
    return __builtin_amdgcn_wmma_f32_16x16x32_bf16(false, a, false, b,
                                                   (short)0, c, false, false);
}

// ===========================================================================
// Kernel 1: QKV projection (x @ w_qkv^T) with fused RoPE epilogue.
//   256 threads = 8 waves (2 x 4); block tile 128(M=s) x 256(N=o); K-step 32.
//   Q -> Qb[head][s][d], K -> Kb[head][s][d], V -> Vt[head][d][s] (transposed)
// ===========================================================================
__global__ __launch_bounds__(256)
void qkv_rope_kernel(const float* __restrict__ x, const float* __restrict__ wqkv,
                     bf16* __restrict__ Qb, bf16* __restrict__ Kb,
                     bf16* __restrict__ Vt)
{
    __shared__ __bf16 As[128 * 32];   //  8 KB
    __shared__ __bf16 Bs[256 * 32];   // 16 KB

    const int tid  = threadIdx.x;
    const int lane = tid & 31;
    const int half = lane >> 4;
    const int l16  = lane & 15;
    const int waveM = (tid >> 5) >> 2;   // 0..1
    const int waveN = (tid >> 5) & 3;    // 0..3
    const int blockM = blockIdx.x * 128; // s
    const int blockN = blockIdx.y * 256; // o

    v8f acc[4][4];
    #pragma unroll
    for (int mf = 0; mf < 4; ++mf)
        #pragma unroll
        for (int nf = 0; nf < 4; ++nf) { v8f z = {}; acc[mf][nf] = z; }

    for (int k0 = 0; k0 < HID; k0 += 32) {
        // Stage A (128x32 f32 -> bf16): 16 elements / thread
        #pragma unroll
        for (int j = 0; j < 16; j += 4) {
            int idx = tid * 16 + j;
            int r = idx >> 5, c = idx & 31;
            v4f f = *(const v4f*)(x + (size_t)(blockM + r) * HID + k0 + c);
            As[r * 32 + c + 0] = (bf16)f[0];
            As[r * 32 + c + 1] = (bf16)f[1];
            As[r * 32 + c + 2] = (bf16)f[2];
            As[r * 32 + c + 3] = (bf16)f[3];
        }
        // Stage B (256x32 f32 -> bf16): rows are w_qkv output features
        #pragma unroll
        for (int j = 0; j < 32; j += 4) {
            int idx = tid * 32 + j;
            int r = idx >> 5, c = idx & 31;
            v4f f = *(const v4f*)(wqkv + (size_t)(blockN + r) * HID + k0 + c);
            Bs[r * 32 + c + 0] = (bf16)f[0];
            Bs[r * 32 + c + 1] = (bf16)f[1];
            Bs[r * 32 + c + 2] = (bf16)f[2];
            Bs[r * 32 + c + 3] = (bf16)f[3];
        }
        __syncthreads();

        v16bf a[4], b[4];
        #pragma unroll
        for (int mf = 0; mf < 4; ++mf)
            a[mf] = load_afrag(&As[(waveM * 64 + mf * 16 + l16) * 32], half);
        #pragma unroll
        for (int nf = 0; nf < 4; ++nf)
            b[nf] = load_bfrag(&Bs[(waveN * 64 + nf * 16 + l16) * 32], half);
        #pragma unroll
        for (int mf = 0; mf < 4; ++mf)
            #pragma unroll
            for (int nf = 0; nf < 4; ++nf)
                acc[mf][nf] = wmma_bf16(a[mf], b[nf], acc[mf][nf]);
        __syncthreads();
    }

    // Epilogue: split into q/k/v, apply RoPE to q,k, store bf16.
    const float kLogTheta = 9.210340371976184f; // ln(10000)
    #pragma unroll
    for (int mf = 0; mf < 4; ++mf) {
        #pragma unroll
        for (int nf = 0; nf < 4; ++nf) {
            int o     = blockN + waveN * 64 + nf * 16 + l16;
            int which = o >> 11;        // 0=q 1=k 2=v (uniform across wave)
            int rem   = o & 2047;
            int head  = rem >> 7;
            int d     = rem & 127;
            #pragma unroll
            for (int i = 0; i < 8; ++i) {
                int s   = blockM + waveM * 64 + mf * 16 + half * 8 + i;
                float v = acc[mf][nf][i];
                if (which < 2) {
                    float other = __shfl_xor(v, 1, 32);   // rotary pair partner
                    float inv_f = __expf(-((float)(d & ~1)) * (kLogTheta / 128.0f));
                    float sn, cs;
                    __sincosf((float)s * inv_f, &sn, &cs);
                    v = ((d & 1) == 0) ? (v * cs - other * sn)
                                       : (other * sn + v * cs);
                }
                bf16 bv = (bf16)v;
                size_t sd = (size_t)head * S_LEN * DH;
                if (which == 0)      Qb[sd + (size_t)s * DH + d] = bv;
                else if (which == 1) Kb[sd + (size_t)s * DH + d] = bv;
                else                 Vt[(size_t)head * DH * S_LEN + (size_t)d * S_LEN + s] = bv;
            }
        }
    }
}

// ===========================================================================
// Kernel 2: causal flash attention. 1 wave / (32-query block, head).
//   QK^T and PV via v_wmma_f32_16x16x32_bf16; online softmax in f32.
// ===========================================================================
__global__ __launch_bounds__(32)
void attn_kernel(const bf16* __restrict__ Qb, const bf16* __restrict__ Kb,
                 const bf16* __restrict__ Vt, bf16* __restrict__ Ob)
{
    __shared__ __bf16 Ps[32 * 32];    // 2 KB P-tile relayout buffer

    const int lane = threadIdx.x & 31;
    const int half = lane >> 4;
    const int l16  = lane & 15;
    const int h    = blockIdx.y;
    const int q0   = blockIdx.x * 32;

    const bf16* Qh = Qb + (size_t)h * S_LEN * DH;
    const bf16* Kh = Kb + (size_t)h * S_LEN * DH;
    const bf16* Vh = Vt + (size_t)h * DH * S_LEN;

    // Q fragments: 32 rows x 128 d, held for the whole key sweep.
    v16bf qf[2][4];
    #pragma unroll
    for (int mf = 0; mf < 2; ++mf)
        #pragma unroll
        for (int kd = 0; kd < 4; ++kd)
            qf[mf][kd] = load_afrag(Qh + (size_t)(q0 + mf * 16 + l16) * DH + kd * 32, half);

    v8f of[2][8];
    #pragma unroll
    for (int mf = 0; mf < 2; ++mf)
        #pragma unroll
        for (int nd = 0; nd < 8; ++nd) { v8f z = {}; of[mf][nd] = z; }

    float mrow[2][8], lrow[2][8];
    #pragma unroll
    for (int mf = 0; mf < 2; ++mf)
        #pragma unroll
        for (int i = 0; i < 8; ++i) { mrow[mf][i] = -1.0e30f; lrow[mf][i] = 0.0f; }

    const float scale = 0.08838834764831845f; // 1/sqrt(128)

    for (int kb = 0; kb <= (int)blockIdx.x; ++kb) {
        const int k0 = kb * 32;

        // ---- S = Q K^T over d (4 x K32 steps) ----
        v8f sacc[2][2];
        #pragma unroll
        for (int mf = 0; mf < 2; ++mf)
            #pragma unroll
            for (int nf = 0; nf < 2; ++nf) { v8f z = {}; sacc[mf][nf] = z; }
        #pragma unroll
        for (int kd = 0; kd < 4; ++kd) {
            v16bf kf0 = load_bfrag(Kh + (size_t)(k0 + l16)      * DH + kd * 32, half);
            v16bf kf1 = load_bfrag(Kh + (size_t)(k0 + 16 + l16) * DH + kd * 32, half);
            sacc[0][0] = wmma_bf16(qf[0][kd], kf0, sacc[0][0]);
            sacc[0][1] = wmma_bf16(qf[0][kd], kf1, sacc[0][1]);
            sacc[1][0] = wmma_bf16(qf[1][kd], kf0, sacc[1][0]);
            sacc[1][1] = wmma_bf16(qf[1][kd], kf1, sacc[1][1]);
        }

        // ---- scale, causal mask, online softmax; P -> LDS (bf16) ----
        #pragma unroll
        for (int mf = 0; mf < 2; ++mf) {
            #pragma unroll
            for (int i = 0; i < 8; ++i) {
                int srow = q0 + mf * 16 + half * 8 + i;
                float a0 = sacc[mf][0][i] * scale;
                float a1 = sacc[mf][1][i] * scale;
                a0 = ((k0 + l16)      <= srow) ? a0 : -1.0e30f;
                a1 = ((k0 + 16 + l16) <= srow) ? a1 : -1.0e30f;
                float mx = fmaxf(a0, a1);
                mx = fmaxf(mx, __shfl_xor(mx, 1, 32));
                mx = fmaxf(mx, __shfl_xor(mx, 2, 32));
                mx = fmaxf(mx, __shfl_xor(mx, 4, 32));
                mx = fmaxf(mx, __shfl_xor(mx, 8, 32));
                float mnew  = fmaxf(mrow[mf][i], mx);
                float alpha = __expf(mrow[mf][i] - mnew);
                float p0 = __expf(a0 - mnew);
                float p1 = __expf(a1 - mnew);
                float ps = p0 + p1;
                ps += __shfl_xor(ps, 1, 32);
                ps += __shfl_xor(ps, 2, 32);
                ps += __shfl_xor(ps, 4, 32);
                ps += __shfl_xor(ps, 8, 32);
                lrow[mf][i] = lrow[mf][i] * alpha + ps;
                mrow[mf][i] = mnew;
                #pragma unroll
                for (int nd = 0; nd < 8; ++nd)
                    of[mf][nd][i] = of[mf][nd][i] * alpha;
                int prow = mf * 16 + half * 8 + i;
                Ps[prow * 32 + l16]      = (bf16)p0;
                Ps[prow * 32 + 16 + l16] = (bf16)p1;
            }
        }
        // LDS is in-order within a wave; fence the DS stores before re-reading.
        asm volatile("s_wait_dscnt 0" ::: "memory");

        // ---- O += P V : reload P as A-fragments, V^T rows are contiguous ----
        v16bf pf[2];
        #pragma unroll
        for (int mf = 0; mf < 2; ++mf)
            pf[mf] = load_afrag(&Ps[(mf * 16 + l16) * 32], half);
        #pragma unroll
        for (int nd = 0; nd < 8; ++nd) {
            v16bf vf = load_bfrag(Vh + (size_t)(nd * 16 + l16) * S_LEN + k0, half);
            of[0][nd] = wmma_bf16(pf[0], vf, of[0][nd]);
            of[1][nd] = wmma_bf16(pf[1], vf, of[1][nd]);
        }
    }

    // ---- normalize + store O as bf16 [s][H] ----
    #pragma unroll
    for (int mf = 0; mf < 2; ++mf) {
        #pragma unroll
        for (int i = 0; i < 8; ++i) {
            float inv = 1.0f / lrow[mf][i];
            int srow  = q0 + mf * 16 + half * 8 + i;
            #pragma unroll
            for (int nd = 0; nd < 8; ++nd) {
                int d = nd * 16 + l16;
                Ob[(size_t)srow * HID + h * DH + d] = (bf16)(of[mf][nd][i] * inv);
            }
        }
    }
}

// ===========================================================================
// Kernel 3: output projection  out = O @ w_out^T.
//   A-tile (128x32 bf16, row-stride 2048) fetched per K-step by the
//   Tensor Data Mover (tensor_load_to_lds, 2D D#), overlapped with the
//   VALU f32->bf16 staging of the B-tile; synced via s_wait_tensorcnt.
// ===========================================================================
__global__ __launch_bounds__(256)
void out_proj_kernel(const bf16* __restrict__ Ob, const float* __restrict__ wout,
                     float* __restrict__ out)
{
    __shared__ __bf16 As[128 * 32];
    __shared__ __bf16 Bs[256 * 32];

    const int tid  = threadIdx.x;
    const int lane = tid & 31;
    const int half = lane >> 4;
    const int l16  = lane & 15;
    const int waveM = (tid >> 5) >> 2;
    const int waveN = (tid >> 5) & 3;
    const int blockM = blockIdx.x * 128;
    const int blockN = blockIdx.y * 256;

    // Only wave 0 drives the TDM (TDM ignores EXEC, so gate with a scalar
    // branch derived from a readfirstlane'd wave id).
    const int wid  = __builtin_amdgcn_readfirstlane(tid >> 5);
    const bool lead = (wid == 0);
    const unsigned lds_a = (unsigned)(uintptr_t)(&As[0]); // low 32b = LDS offset

    v8f acc[4][4];
    #pragma unroll
    for (int mf = 0; mf < 4; ++mf)
        #pragma unroll
        for (int nf = 0; nf < 4; ++nf) { v8f z = {}; acc[mf][nf] = z; }

    for (int k0 = 0; k0 < HID; k0 += 32) {
        // ---- A tile via Tensor Data Mover: 128x32 bf16 tile of Ob ----
        if (lead) {
            unsigned long long ga =
                (unsigned long long)(uintptr_t)(Ob + (size_t)blockM * HID + k0);
            v4u g0;
            g0[0] = 1u;                                   // count=1 (user D#)
            g0[1] = lds_a;                                // lds_addr
            g0[2] = (unsigned)(ga & 0xFFFFFFFFu);         // global_addr[31:0]
            g0[3] = (unsigned)((ga >> 32) & 0x01FFFFFFu)  // global_addr[56:32]
                    | (2u << 30);                         // type=2 ("image")
            v8u g1;
            g1[0] = (1u << 16);          // workgroup_mask=0, data_size=1 (2B)
            g1[1] = (2048u << 16);       // tensor_dim0[15:0]=2048
            g1[2] = (2048u << 16);       // tensor_dim0[31:16]=0, tensor_dim1[15:0]=2048
            g1[3] = (32u << 16);         // tensor_dim1[31:16]=0, tile_dim0=32
            g1[4] = 128u;                // tile_dim1=128, tile_dim2=0
            g1[5] = 2048u;               // tensor_dim0_stride[31:0]=2048
            g1[6] = 0u;                  // stride0[47:32]=0, stride1[15:0]=0
            g1[7] = 0u;                  // stride1[47:16]=0
            asm volatile("tensor_load_to_lds %0, %1" :: "s"(g0), "s"(g1) : "memory");
        }
        // ---- B tile staged by all threads (f32 -> bf16), overlaps the DMA ----
        #pragma unroll
        for (int j = 0; j < 32; j += 4) {
            int idx = tid * 32 + j;
            int r = idx >> 5, c = idx & 31;
            v4f f = *(const v4f*)(wout + (size_t)(blockN + r) * HID + k0 + c);
            Bs[r * 32 + c + 0] = (bf16)f[0];
            Bs[r * 32 + c + 1] = (bf16)f[1];
            Bs[r * 32 + c + 2] = (bf16)f[2];
            Bs[r * 32 + c + 3] = (bf16)f[3];
        }
        if (lead) __builtin_amdgcn_s_wait_tensorcnt(0);
        __syncthreads();

        v16bf a[4], b[4];
        #pragma unroll
        for (int mf = 0; mf < 4; ++mf)
            a[mf] = load_afrag(&As[(waveM * 64 + mf * 16 + l16) * 32], half);
        #pragma unroll
        for (int nf = 0; nf < 4; ++nf)
            b[nf] = load_bfrag(&Bs[(waveN * 64 + nf * 16 + l16) * 32], half);
        #pragma unroll
        for (int mf = 0; mf < 4; ++mf)
            #pragma unroll
            for (int nf = 0; nf < 4; ++nf)
                acc[mf][nf] = wmma_bf16(a[mf], b[nf], acc[mf][nf]);
        __syncthreads();
    }

    #pragma unroll
    for (int mf = 0; mf < 4; ++mf)
        #pragma unroll
        for (int nf = 0; nf < 4; ++nf) {
            int o = blockN + waveN * 64 + nf * 16 + l16;
            #pragma unroll
            for (int i = 0; i < 8; ++i) {
                int s = blockM + waveM * 64 + mf * 16 + half * 8 + i;
                out[(size_t)s * HID + o] = acc[mf][nf][i];
            }
        }
}

// ===========================================================================
extern "C" void kernel_launch(void* const* d_in, const int* in_sizes, int n_in,
                              void* d_out, int out_size, void* d_ws, size_t ws_size,
                              hipStream_t stream) {
    const float* x    = (const float*)d_in[0];
    const float* wqkv = (const float*)d_in[1];
    const float* wout = (const float*)d_in[2];
    float* out = (float*)d_out;

    char* ws = (char*)d_ws;
    const size_t MB8 = (size_t)8 * 1024 * 1024;
    bf16* Qb = (bf16*)(ws + 0 * MB8);  // [16][2048][128]
    bf16* Kb = (bf16*)(ws + 1 * MB8);  // [16][2048][128]
    bf16* Vt = (bf16*)(ws + 2 * MB8);  // [16][128][2048]  (transposed)
    bf16* Ob = (bf16*)(ws + 3 * MB8);  // [2048][2048]

    dim3 g1(S_LEN / 128, O3 / 256);    // 16 x 24
    qkv_rope_kernel<<<g1, 256, 0, stream>>>(x, wqkv, Qb, Kb, Vt);

    dim3 g2(S_LEN / 32, NH);           // 64 x 16
    attn_kernel<<<g2, 32, 0, stream>>>(Qb, Kb, Vt, Ob);

    dim3 g3(S_LEN / 128, HID / 256);   // 16 x 8
    out_proj_kernel<<<g3, 256, 0, stream>>>(Ob, wout, out);
}
